// GCNResNetLayer_15753940042116
// MI455X (gfx1250) — matrix operand order repaired and checked by reading, not compiled, not run on hardware
//
#include <hip/hip_runtime.h>

typedef __attribute__((ext_vector_type(16))) _Float16     v16h;
typedef __attribute__((ext_vector_type(8)))  float        v8f;
typedef __attribute__((ext_vector_type(4)))  unsigned int v4u;
typedef __attribute__((ext_vector_type(8)))  unsigned int v8u;

union ABfrag { v4u u[2]; v16h h; };

#define IMG_ELEMS 65536   // 64 ch * 1024 px
#define PX        1024
#define CCH       64
#define WELEMS    36864   // 64*64*9

// LDS byte-offset of a generic pointer that is known to point into LDS.
typedef __attribute__((address_space(3))) const char* lds_cptr_t;
__device__ __forceinline__ unsigned lds_off(const void* p) {
  return (unsigned)(uintptr_t)(lds_cptr_t)p;
}

__device__ __forceinline__ void async_copy_b128(unsigned lds_byte_off, const void* gaddr) {
  // CDNA5: DMA 16B global -> LDS, no VGPR round trip, tracked by ASYNCcnt.
  asm volatile("global_load_async_to_lds_b128 %0, %1, off"
               :: "v"(lds_byte_off), "v"((unsigned long long)(uintptr_t)gaddr)
               : "memory");
}

__device__ __forceinline__ void wait_asynccnt0() {
#if __has_builtin(__builtin_amdgcn_s_wait_asynccnt)
  __builtin_amdgcn_s_wait_asynccnt(0);
#else
  asm volatile("s_wait_asynccnt 0" ::: "memory");
#endif
}

// ---------------------------------------------------------------------------
// TDM: one tensor_load_to_lds moving `n8` 8-byte units (contiguous 1-D tile)
// from global to LDS. 2-group D# form (VADDR2/3 disabled -> tensors up to 2D).
// Bitfields per CDNA5 ISA ch.8 (D# Group 0 / Group 1).
// ---------------------------------------------------------------------------
__device__ __forceinline__ void tdm_load_1d(unsigned lds_byte_off,
                                            const void* gptr, unsigned n8) {
  const unsigned long long ga = (unsigned long long)(uintptr_t)gptr;
  v4u g0;
  g0.x = 1u;                                                  // count=1 (valid), user mode
  g0.y = lds_byte_off;                                        // lds_addr [63:32]
  g0.z = (unsigned)ga;                                        // global_addr[31:0]
  g0.w = ((unsigned)(ga >> 32) & 0x01ffffffu) | 0x80000000u;  // addr[56:32] | type=2
  v8u g1;
  g1[0] = 3u << 16;                            // data_size = 8B; no flags, no mask
  g1[1] = (n8 & 0xffffu) << 16;                // [63:48] tensor_dim0[15:0]
  g1[2] = ((n8 >> 16) & 0xffffu) | (1u << 16); // tensor_dim0[31:16] | tensor_dim1 = 1
  g1[3] = (n8 & 0xffffu) << 16;                // tensor_dim1 hi = 0 | tile_dim0 = n8
  g1[4] = 1u;                                  // tile_dim1 = 1, tile_dim2 = 0
  g1[5] = n8;                                  // tensor_dim0_stride[31:0]
  g1[6] = 0u;                                  // stride0[47:32] | dim1_stride[15:0]
  g1[7] = 0u;                                  // dim1_stride[47:16]
  asm volatile("tensor_load_to_lds %0, %1" :: "s"(g0), "s"(g1) : "memory");
}

__device__ __forceinline__ void wait_tensorcnt0() {
#if __has_builtin(__builtin_amdgcn_s_wait_tensorcnt)
  __builtin_amdgcn_s_wait_tensorcnt(0);
#else
  asm volatile("s_wait_tensorcnt 0" ::: "memory");
#endif
}

// ---------------------------------------------------------------------------
// Pack f32 image [n][ci][p] -> f16 [n][p][ci] (the conv kernel's DMA format).
// ---------------------------------------------------------------------------
__global__ __launch_bounds__(256) void pack_x_f16(const float* __restrict__ src,
                                                  _Float16* __restrict__ dst) {
  const int n = blockIdx.x;
  const int p = blockIdx.y * 256 + threadIdx.x;
  const float*   s = src + (size_t)n * IMG_ELEMS + p;
  _Float16*      d = dst + (size_t)n * IMG_ELEMS + (size_t)p * CCH;
  union { v4u u; _Float16 h[8]; } pk;
#pragma unroll
  for (int c0 = 0; c0 < CCH; c0 += 8) {
#pragma unroll
    for (int j = 0; j < 8; ++j) pk.h[j] = (_Float16)s[(size_t)(c0 + j) * PX];
    *(v4u*)(d + c0) = pk.u;
  }
}

// ---------------------------------------------------------------------------
// Pack all 4 weight tensors (co,ci,ky,kx) f32 -> f16 [t][shift][co][ci].
// ---------------------------------------------------------------------------
__global__ __launch_bounds__(256) void pack_w_f16(
    const float* __restrict__ w0, const float* __restrict__ w1,
    const float* __restrict__ w2, const float* __restrict__ w3,
    _Float16* __restrict__ dst) {
  int i = blockIdx.x * 256 + threadIdx.x;
  if (i >= 4 * WELEMS) return;
  int t = i / WELEMS;
  int r = i - t * WELEMS;
  const float* w = (t == 0) ? w0 : (t == 1) ? w1 : (t == 2) ? w2 : w3;
  int co  = r / 576;
  int rem = r - co * 576;
  int ci  = rem / 9;
  int s   = rem - ci * 9;
  dst[(size_t)t * WELEMS + ((s * CCH + co) * CCH + ci)] = (_Float16)w[r];
}

// ---------------------------------------------------------------------------
// Fused dual 3x3 conv. One workgroup per image.
// Prologue: image (128KB) staged by ONE tensor_load_to_lds TDM op (wave 0),
// weights (144KB) staged by per-lane global_load_async_to_lds_b128 across all
// waves; s_wait_tensorcnt / s_wait_asynccnt + barrier. Main loop: 9 shifted
// 1x1 GEMMs, K=64 in two k-steps, v_wmma_f32_16x16x32_f16 on 16co x 16px tiles.
// ---------------------------------------------------------------------------
__global__ __launch_bounds__(256) void conv3x3_dual_wmma(
    const _Float16* __restrict__ xpack,   // [n][p][ci] f16
    const _Float16* __restrict__ wpackL,  // [2][9][co][ci] f16 (x then n branch)
    const float* __restrict__ bx, const float* __restrict__ bn,
    float* __restrict__ outx, float* __restrict__ outn)
{
  extern __shared__ _Float16 smem[];
  _Float16* xs = smem;               // [1024][64]     128 KB
  _Float16* ws = smem + IMG_ELEMS;   // [2][9][64][64] 144 KB

  const int n   = blockIdx.x;
  const int tid = threadIdx.x;

  // ---- weights: async DMA spread over all waves ----
  {
    const char* wg = (const char*)wpackL;
    const unsigned wbase0 = lds_off(ws);
    for (int k = tid; k < 2 * WELEMS * 2 / 16; k += 256)          // 9216 x 16B
      async_copy_b128(wbase0 + k * 16, wg + k * 16);
  }
  // ---- image: single TDM op from wave 0 (EXEC-independent, per-wave issue) ----
  if (tid < 32) {
    tdm_load_1d(lds_off(xs), xpack + (size_t)n * IMG_ELEMS,
                IMG_ELEMS * 2 / 8 /* = 16384 8B units = 128KB */);
    wait_tensorcnt0();
  }
  wait_asynccnt0();
  __syncthreads();

  const int wave  = tid >> 5;
  const int lane  = tid & 31;
  const int laneN = lane & 15;
  const int hi    = lane >> 4;   // half-wave selector per ISA fragment layout

  // 512 output tiles (2 branches * 4 co-tiles * 64 pixel-tiles) over 8 waves
  for (int t = wave; t < 512; t += 8) {
    const int branch = t >> 8;
    const int tt = t & 255;
    const int mt = tt >> 6;     // co tile 0..3
    const int nt = tt & 63;     // pixel tile 0..63

    const _Float16* wb = ws + branch * WELEMS;
    const int co  = mt * 16 + laneN;   // A: M = lane&15
    const int p   = nt * 16 + laneN;   // B: N = lane&15
    const int py  = p >> 5;
    const int pxc = p & 31;

    v8f acc = {0.f, 0.f, 0.f, 0.f, 0.f, 0.f, 0.f, 0.f};

    for (int s = 0; s < 9; ++s) {
      const int dy = s / 3 - 1;
      const int dx = s - (s / 3) * 3 - 1;
      const int yy = py + dy;
      const int xx = pxc + dx;
      const bool inb = (yy >= 0) & (yy < 32) & (xx >= 0) & (xx < 32);
      const int pp = yy * 32 + xx;
      const _Float16* arow = wb + (s * CCH + co) * CCH;
#pragma unroll
      for (int k0 = 0; k0 < CCH; k0 += 32) {
        ABfrag a, b;
        // A 16x32 f16: lanes<16 K = k0+{0..7,16..23}; lanes>=16 K = k0+{8..15,24..31}
        const _Float16* ap = arow + k0 + hi * 8;
        a.u[0] = *(const v4u*)(ap);
        a.u[1] = *(const v4u*)(ap + 16);
        // B 32x16 f16: lanes<16 K = k0+{0..7,16..23}; lanes>=16 K = k0+{8..15,24..31} (N = lane&15)
        if (inb) {
          const _Float16* bp = xs + pp * CCH + k0 + hi * 8;
          b.u[0] = *(const v4u*)(bp);
          b.u[1] = *(const v4u*)(bp + 16);
        } else {
          v4u z = {0u, 0u, 0u, 0u};
          b.u[0] = z; b.u[1] = z;   // zero-pad border; EXEC stays all-1s
        }
        acc = __builtin_amdgcn_wmma_f32_16x16x32_f16(
            false, a.h, false, b.h, (short)0, acc, false, false);
      }
    }

    // D layout: lanes<16 -> M=r; lanes>=16 -> M=r+8; N = lane&15.
    float* outp = branch ? outn : outx;
    const float* bias = branch ? bn : bx;
    const int coBase = mt * 16 + hi * 8;
    float* ob = outp + ((size_t)n * CCH + coBase) * PX + (nt * 16 + laneN);
#pragma unroll
    for (int r = 0; r < 8; ++r)
      ob[r * PX] = acc[r] + bias[coBase + r];
  }
}

// ---------------------------------------------------------------------------
// h = relu(hx + sum_{j in neighbors(b,i)} nset[b,j] + optional residual)
// ---------------------------------------------------------------------------
__global__ __launch_bounds__(256) void nsum_combine_relu(
    const float* __restrict__ hx, const float* __restrict__ nset,
    const unsigned char* __restrict__ mask, const float* __restrict__ residual,
    float* __restrict__ out)
{
  __shared__ int nidx[64];
  __shared__ int ncnt;
  const int bi  = blockIdx.x;      // b*64 + i
  const int b   = bi >> 6;
  const int tid = threadIdx.x;

  if (tid == 0) {
    const unsigned char* mrow = mask + (size_t)bi * 64;
    int c = 0;
    for (int j = 0; j < 64; ++j)
      if (mrow[j]) nidx[c++] = j;
    ncnt = c;
  }
  __syncthreads();
  const int cn = ncnt;

  const size_t e    = (size_t)blockIdx.y * 1024 + (size_t)tid * 4;
  const size_t base = (size_t)bi * IMG_ELEMS + e;

  float4 v = *(const float4*)(hx + base);
  for (int k = 0; k < cn; ++k) {
    const float4 nv = *(const float4*)(nset + (size_t)(b * 64 + nidx[k]) * IMG_ELEMS + e);
    v.x += nv.x; v.y += nv.y; v.z += nv.z; v.w += nv.w;
  }
  if (residual) {
    const float4 rv = *(const float4*)(residual + base);
    v.x += rv.x; v.y += rv.y; v.z += rv.z; v.w += rv.w;
  }
  v.x = v.x > 0.f ? v.x : 0.f;
  v.y = v.y > 0.f ? v.y : 0.f;
  v.z = v.z > 0.f ? v.z : 0.f;
  v.w = v.w > 0.f ? v.w : 0.f;
  *(float4*)(out + base) = v;
}

// ---------------------------------------------------------------------------
extern "C" void kernel_launch(void* const* d_in, const int* in_sizes, int n_in,
                              void* d_out, int out_size, void* d_ws, size_t ws_size,
                              hipStream_t stream) {
  const float* x    = (const float*)d_in[0];
  const float* w_x0 = (const float*)d_in[1];
  const float* b_x0 = (const float*)d_in[2];
  const float* w_n0 = (const float*)d_in[3];
  const float* b_n0 = (const float*)d_in[4];
  const float* w_x1 = (const float*)d_in[5];
  const float* b_x1 = (const float*)d_in[6];
  const float* w_n1 = (const float*)d_in[7];
  const float* b_n1 = (const float*)d_in[8];
  const unsigned char* mask = (const unsigned char*)d_in[9];
  float* out = (float*)d_out;

  const size_t IMG_TOTAL = (size_t)256 * IMG_ELEMS;   // 16,777,216 elements
  float*    buf0  = (float*)d_ws;                     // conv_x output (f32)
  float*    buf1  = buf0 + IMG_TOTAL;                 // conv_n output (f32)
  float*    buf2  = buf1 + IMG_TOTAL;                 // h activation  (f32)
  _Float16* xpack = (_Float16*)(buf2 + IMG_TOTAL);    // f16 [n][p][ci]
  _Float16* wpack = xpack + IMG_TOTAL;                // f16 [4][9][co][ci]

  const size_t smem = (size_t)(IMG_ELEMS + 2 * WELEMS) * sizeof(_Float16); // 272 KB

  pack_w_f16<<<(4 * WELEMS + 255) / 256, 256, 0, stream>>>(w_x0, w_n0, w_x1, w_n1, wpack);
  pack_x_f16<<<dim3(256, 4), 256, 0, stream>>>(x, xpack);
  conv3x3_dual_wmma<<<256, 256, smem, stream>>>(xpack, wpack, b_x0, b_n0, buf0, buf1);
  nsum_combine_relu<<<dim3(256, 64), 256, 0, stream>>>(buf0, buf1, mask, nullptr, buf2);
  pack_x_f16<<<dim3(256, 4), 256, 0, stream>>>(buf2, xpack);
  conv3x3_dual_wmma<<<256, 256, smem, stream>>>(xpack, wpack + (size_t)2 * WELEMS,
                                                b_x1, b_n1, buf0, buf1);
  nsum_combine_relu<<<dim3(256, 64), 256, 0, stream>>>(buf0, buf1, mask, x, out);
}
